// Block_39247411151159
// MI455X (gfx1250) — matrix operand order repaired and checked by reading, not compiled, not run on hardware
//
#include <hip/hip_runtime.h>

// ---------------------------------------------------------------------------
// CDNA5 (gfx1250) wave32 BF16 WMMA implementation of the external-memory
// attention block.  LDS-staged GEMM: global_load_b128 f32 loads ->
// v_cvt_pk_bf16_f32 -> ds_store_b64 -> ds_load_b128 fragments ->
// v_wmma_f32_16x16x32_bf16, 2x2 WMMA tiles per wave (f32 accumulation).
// ---------------------------------------------------------------------------

typedef __attribute__((ext_vector_type(16))) __bf16 v16bf;
typedef __attribute__((ext_vector_type(8)))  __bf16 v8bf;
typedef __attribute__((ext_vector_type(2)))  __bf16 v2bf;
typedef __attribute__((ext_vector_type(8)))  float  v8f;

static __device__ __forceinline__ unsigned short bf16u(float f) {
    __bf16 b = (__bf16)f;                 // native f32->bf16 (RNE)
    unsigned short u;
    __builtin_memcpy(&u, &b, 2);
    return u;
}
static __device__ __forceinline__ unsigned pack2(float lo, float hi) {
    // lowers to a single v_cvt_pk_bf16_f32
    v2bf v; v[0] = (__bf16)lo; v[1] = (__bf16)hi;
    unsigned u;
    __builtin_memcpy(&u, &v, 4);
    return u;
}

// Generic strided GEMM:  C[m,n] = sum_k (A(m,k) * kscale?[k] * sscale?) * B(k,n)
//                        (+ bias?[m])
// A(m,k) at A[m*sa_m + k*sa_k], B(k,n) at B[k*sb_k + n*sb_n], C row stride ldc.
// Template: KS  = kscale present,
//           AKU = sa_k==1 (else sa_m==1 assumed),
//           BNU = sb_n==1 (else sb_k==1 assumed).
// Block = 256 threads = 8 waves.  Block tile 128(M) x 64(N), k-step 32.
// Wave tile 32x32 = 2x2 WMMA tiles (4 accumulators).
// Requires M%128==0, N%64==0, K%32==0 and 16B-aligned operand slices.
template <bool KS, bool AKU, bool BNU>
__global__ __launch_bounds__(256)
void gemm_bf16_wmma(float* __restrict__ C,
                    const float* __restrict__ A,
                    const float* __restrict__ B,
                    const float* __restrict__ bias,
                    const float* __restrict__ kscale,
                    const float* __restrict__ sscale,
                    int M, int N, int K,
                    long sa_m, long sa_k, long sb_k, long sb_n, long ldc) {
    constexpr int LSTR = 40;                      // row stride in bf16: 32 data + pad (80B, 16B-aligned)
    __shared__ unsigned short As[128 * LSTR];     // As[i][k]  (m-major)
    __shared__ unsigned short Bs[64 * LSTR];      // Bs[n][k]  (transposed in LDS)

    const int t    = (int)threadIdx.x;
    const int wave = t >> 5;
    const int lane = t & 31;
    const int wm   = wave & 3;                    // 32-row band 0..3
    const int wn   = wave >> 2;                   // 32-col band 0..1
    const int hl   = lane >> 4;                   // lane half
    const int idx  = lane & 15;                   // row (A) / col (B,C) within tile

    const int m0 = (int)blockIdx.y * 128;
    const int n0 = (int)blockIdx.x * 64;

    const float ss = sscale ? *sscale : 1.0f;

    const float* __restrict__ Ablk = A + (long)m0 * sa_m;
    const float* __restrict__ Bblk = B + (long)n0 * sb_n;

    v8f acc00 = {}, acc01 = {}, acc10 = {}, acc11 = {};

    for (int k0 = 0; k0 < K; k0 += 32) {
        __syncthreads();                          // LDS reuse across k-steps

        // ---- stage A tile (128 x 32) into As[i*LSTR + k], folding scales ----
        if (AKU) {                                // unit stride along k: float4 + ds_store_b64
#pragma unroll
            for (int it = 0; it < 4; ++it) {
                int c = it * 256 + t;             // 1024 float4 chunks
                int i = c >> 3;
                int k = (c & 7) * 4;
                float4 a4 = *(const float4*)(Ablk + (long)i * sa_m + (k0 + k));
                float x0 = a4.x, x1 = a4.y, x2 = a4.z, x3 = a4.w;
                if (KS) {
                    float4 k4 = *(const float4*)(kscale + k0 + k);
                    x0 *= k4.x; x1 *= k4.y; x2 *= k4.z; x3 *= k4.w;
                }
                uint2 p; p.x = pack2(x0 * ss, x1 * ss); p.y = pack2(x2 * ss, x3 * ss);
                *(uint2*)(As + i * LSTR + k) = p;
                if (it == 0 && k0 + 32 < K)
                    __builtin_prefetch(Ablk + (long)i * sa_m + (k0 + 32 + k), 0, 1);
            }
        } else {                                  // unit stride along m: float4 + transpose scatter
#pragma unroll
            for (int it = 0; it < 4; ++it) {
                int c = it * 256 + t;
                int k = c >> 5;
                int i = (c & 31) * 4;
                float4 a4 = *(const float4*)(Ablk + (long)i * sa_m + (long)(k0 + k) * sa_k);
                float sk = ss;
                if (KS) sk *= kscale[k0 + k];
                As[(i + 0) * LSTR + k] = bf16u(a4.x * sk);
                As[(i + 1) * LSTR + k] = bf16u(a4.y * sk);
                As[(i + 2) * LSTR + k] = bf16u(a4.z * sk);
                As[(i + 3) * LSTR + k] = bf16u(a4.w * sk);
                if (it == 0 && k0 + 32 < K)
                    __builtin_prefetch(Ablk + (long)i * sa_m + (long)(k0 + 32 + k) * sa_k, 0, 1);
            }
        }
        // ---- stage B tile (32 x 64) transposed into Bs[n*LSTR + k] ----
        if (BNU) {                                // unit stride along n: float4 + transpose scatter
#pragma unroll
            for (int it = 0; it < 2; ++it) {
                int c = it * 256 + t;             // 512 float4 chunks
                int k = c >> 4;
                int j = (c & 15) * 4;
                float4 b4 = *(const float4*)(Bblk + (long)(k0 + k) * sb_k + j);
                Bs[(j + 0) * LSTR + k] = bf16u(b4.x);
                Bs[(j + 1) * LSTR + k] = bf16u(b4.y);
                Bs[(j + 2) * LSTR + k] = bf16u(b4.z);
                Bs[(j + 3) * LSTR + k] = bf16u(b4.w);
                if (it == 0 && k0 + 32 < K)
                    __builtin_prefetch(Bblk + (long)(k0 + 32 + k) * sb_k + j, 0, 1);
            }
        } else {                                  // unit stride along k: float4 + ds_store_b64
#pragma unroll
            for (int it = 0; it < 2; ++it) {
                int c = it * 256 + t;
                int j = c >> 3;
                int k = (c & 7) * 4;
                float4 b4 = *(const float4*)(Bblk + (long)j * sb_n + (k0 + k));
                uint2 p; p.x = pack2(b4.x, b4.y); p.y = pack2(b4.z, b4.w);
                *(uint2*)(Bs + j * LSTR + k) = p;
                if (it == 0 && k0 + 32 < K)
                    __builtin_prefetch(Bblk + (long)j * sb_n + (k0 + 32 + k), 0, 1);
            }
        }
        __syncthreads();

        // ---- A fragments (16-bit A 16x32 layout, ISA 7.12.2):
        //      lane half 0: K 0..7 / 16..23 ; half 1: K 8..15 / 24..31
        v16bf af0, af1;
        {
            const unsigned short* ap = As + (wm * 32 + idx) * LSTR;
            v8bf lo = *(const v8bf*)(ap + hl * 8);        // ds_load_b128
            v8bf hi = *(const v8bf*)(ap + 16 + hl * 8);   // ds_load_b128
#pragma unroll
            for (int j = 0; j < 8; ++j) { af0[j] = lo[j]; af0[j + 8] = hi[j]; }
        }
        {
            const unsigned short* ap = As + (wm * 32 + 16 + idx) * LSTR;
            v8bf lo = *(const v8bf*)(ap + hl * 8);
            v8bf hi = *(const v8bf*)(ap + 16 + hl * 8);
#pragma unroll
            for (int j = 0; j < 8; ++j) { af1[j] = lo[j]; af1[j + 8] = hi[j]; }
        }
        // ---- B fragments: lane holds column n, K = hl*8 + j / 16 + hl*8 + j ----
        v16bf bf0, bf1;
        {
            const unsigned short* bp = Bs + (wn * 32 + idx) * LSTR + hl * 8;
            v8bf lo = *(const v8bf*)(bp);
            v8bf hi = *(const v8bf*)(bp + 16);
#pragma unroll
            for (int j = 0; j < 8; ++j) { bf0[j] = lo[j]; bf0[j + 8] = hi[j]; }
        }
        {
            const unsigned short* bp = Bs + (wn * 32 + 16 + idx) * LSTR + hl * 8;
            v8bf lo = *(const v8bf*)(bp);
            v8bf hi = *(const v8bf*)(bp + 16);
#pragma unroll
            for (int j = 0; j < 8; ++j) { bf1[j] = lo[j]; bf1[j + 8] = hi[j]; }
        }

        acc00 = __builtin_amdgcn_wmma_f32_16x16x32_bf16(false, af0, false, bf0, (short)0, acc00, false, false);
        acc01 = __builtin_amdgcn_wmma_f32_16x16x32_bf16(false, af0, false, bf1, (short)0, acc01, false, false);
        acc10 = __builtin_amdgcn_wmma_f32_16x16x32_bf16(false, af1, false, bf0, (short)0, acc10, false, false);
        acc11 = __builtin_amdgcn_wmma_f32_16x16x32_bf16(false, af1, false, bf1, (short)0, acc11, false, false);
    }

    // ---- epilogue: C/D layout -> lane column idx, rows hl*8 + r ----
    const int cn = n0 + wn * 32 + idx;
    if (bias) {
#pragma unroll
        for (int r = 0; r < 8; ++r) {
            int ma = m0 + wm * 32 + hl * 8 + r;
            int mb = ma + 16;
            float bva = bias[ma], bvb = bias[mb];
            C[(long)ma * ldc + cn]      = acc00[r] + bva;
            C[(long)ma * ldc + cn + 16] = acc01[r] + bva;
            C[(long)mb * ldc + cn]      = acc10[r] + bvb;
            C[(long)mb * ldc + cn + 16] = acc11[r] + bvb;
        }
    } else {
#pragma unroll
        for (int r = 0; r < 8; ++r) {
            int ma = m0 + wm * 32 + hl * 8 + r;
            int mb = ma + 16;
            C[(long)ma * ldc + cn]      = acc00[r];
            C[(long)ma * ldc + cn + 16] = acc01[r];
            C[(long)mb * ldc + cn]      = acc10[r];
            C[(long)mb * ldc + cn + 16] = acc11[r];
        }
    }
}

// Softmax over the E (=512) axis of g[b*H+h, e, s] with per-column temperature
// scale = mask[b,s] * regular / sqrt_p applied BEFORE exp.  One wave per column.
__global__ __launch_bounds__(256)
void softmax_etc(float* __restrict__ g,
                 const float* __restrict__ mask,
                 const float* __restrict__ regular,
                 const float* __restrict__ sqrt_p) {
    const int E = 512, S = 1024;
    const int wave = (int)threadIdx.x >> 5;
    const int lane = (int)threadIdx.x & 31;
    const int s  = (int)blockIdx.x * 8 + wave;
    const int bh = (int)blockIdx.y;
    const int b  = bh >> 3;                    // H == 8

    const float scale = mask[b * S + s] * (*regular) / (*sqrt_p);
    float* col = g + (long)bh * E * S + s;

    float x[16];
    float mx = -3.402823466e38f;
#pragma unroll
    for (int i = 0; i < 16; ++i) {
        x[i] = col[(long)(i * 32 + lane) * S] * scale;
        mx = fmaxf(mx, x[i]);
    }
#pragma unroll
    for (int off = 16; off > 0; off >>= 1) mx = fmaxf(mx, __shfl_xor(mx, off, 32));
    float sum = 0.0f;
#pragma unroll
    for (int i = 0; i < 16; ++i) { x[i] = __expf(x[i] - mx); sum += x[i]; }
#pragma unroll
    for (int off = 16; off > 0; off >>= 1) sum += __shfl_xor(sum, off, 32);
    const float inv = 1.0f / sum;
#pragma unroll
    for (int i = 0; i < 16; ++i) col[(long)(i * 32 + lane) * S] = x[i] * inv;
}

// out1[b, h*D+d, s] += avgpool3_same(multi_q) where
// multi_q[b,h,d,s] = q[b,d,s] * mix[h,d] * mask[b,s] * regular
__global__ __launch_bounds__(256)
void pool_add(float* __restrict__ out1,
              const float* __restrict__ q,
              const float* __restrict__ mix,
              const float* __restrict__ mask,
              const float* __restrict__ regular) {
    const int S = 1024, D = 256, H = 8, B = 4;
    long idx = (long)blockIdx.x * blockDim.x + (long)threadIdx.x;
    if (idx >= (long)B * H * D * S) return;
    int s = (int)(idx & (S - 1));
    long tt = idx >> 10;
    int d = (int)(tt & (D - 1)); tt >>= 8;
    int h = (int)(tt & (H - 1));
    int b = (int)(tt >> 3);

    const float mixv = mix[h * D + d];
    const float reg  = *regular;
    const float* qrow = q + ((long)b * D + d) * S;
    const float* mrow = mask + (long)b * S;

    float acc = 0.0f;
#pragma unroll
    for (int o = -1; o <= 1; ++o) {
        int sp = s + o;
        if (sp >= 0 && sp < S) acc += qrow[sp] * mrow[sp];
    }
    acc *= mixv * reg * (1.0f / 3.0f);
    out1[(((long)b * H + h) * D + d) * S + s] += acc;
}

// Host-side dispatcher picks the right template instantiation.
static void launch_gemm(dim3 grid, hipStream_t stream,
                        float* C, const float* A, const float* B,
                        const float* bias, const float* kscale, const float* sscale,
                        int M, int N, int K,
                        long sa_m, long sa_k, long sb_k, long sb_n, long ldc) {
    dim3 blk(256);
    const bool ks  = (kscale != nullptr);
    const bool aku = (sa_k == 1);
    const bool bnu = (sb_n == 1);
#define LG(KS, AK, BN) gemm_bf16_wmma<KS, AK, BN><<<grid, blk, 0, stream>>>( \
        C, A, B, bias, kscale, sscale, M, N, K, sa_m, sa_k, sb_k, sb_n, ldc)
    if (ks) {
        if (aku) { if (bnu) LG(true, true, true);  else LG(true, true, false); }
        else     { if (bnu) LG(true, false, true); else LG(true, false, false); }
    } else {
        if (aku) { if (bnu) LG(false, true, true);  else LG(false, true, false); }
        else     { if (bnu) LG(false, false, true); else LG(false, false, false); }
    }
#undef LG
}

extern "C" void kernel_launch(void* const* d_in, const int* in_sizes, int n_in,
                              void* d_out, int out_size, void* d_ws, size_t ws_size,
                              hipStream_t stream) {
    (void)in_sizes; (void)n_in; (void)out_size; (void)ws_size;
    const float* y       = (const float*)d_in[0];   // [B, D, S]
    const float* e_s     = (const float*)d_in[1];   // [H, B, E, S]
    const float* mask    = (const float*)d_in[2];   // [B, 1, S]
    const float* regular = (const float*)d_in[3];   // scalar
    const float* mix     = (const float*)d_in[4];   // [H, D, 1]
    const float* sqrt_p  = (const float*)d_in[5];   // scalar
    const float* q_w     = (const float*)d_in[6];   // [D, D]
    const float* q_b     = (const float*)d_in[7];   // [D]
    const float* v_w     = (const float*)d_in[8];   // [H*D, D]
    const float* v_b     = (const float*)d_in[9];   // [H*D]
    const float* re_w    = (const float*)d_in[10];  // [D, H*D]
    const float* re_b    = (const float*)d_in[11];  // [D]
    const float* sq_w    = (const float*)d_in[12];  // [D, H*D]
    const float* sq_b    = (const float*)d_in[13];  // [D]

    const int B = 4, H = 8, D = 256, E = 512, S = 1024, HD = 2048;

    float* ws   = (float*)d_ws;
    float* q_ws = ws;  ws += (size_t)B * D * S;        //  4 MB
    float* v_ws = ws;  ws += (size_t)B * HD * S;       // 32 MB
    float* etcv = ws;  ws += (size_t)B * H * D * E;    // 16 MB
    float* etck = ws;  ws += (size_t)B * D * E;        //  2 MB
    float* sc   = ws;  ws += (size_t)B * H * E * S;    // 64 MB
    float* out1 = ws;  ws += (size_t)B * HD * S;       // 32 MB
    float* outF = (float*)d_out;                       // [B, D, S]

    // 1) q[b] = q_w @ y[b] + q_b          (M=D, N=S, K=D)
    for (int b = 0; b < B; ++b)
        launch_gemm(dim3(S / 64, D / 128), stream,
                    q_ws + (size_t)b * D * S, q_w, y + (size_t)b * D * S,
                    q_b, nullptr, nullptr,
                    D, S, D, /*sa_m*/D, /*sa_k*/1, /*sb_k*/S, /*sb_n*/1, /*ldc*/S);

    // 2) v[b] = v_w @ y[b] + v_b          (M=HD, N=S, K=D)
    for (int b = 0; b < B; ++b)
        launch_gemm(dim3(S / 64, HD / 128), stream,
                    v_ws + (size_t)b * HD * S, v_w, y + (size_t)b * D * S,
                    v_b, nullptr, nullptr,
                    HD, S, D, D, 1, S, 1, S);

    // 3) etc_v[b,h][d,e] = sum_s vh[b,h,d,s]*(h==0 ? mask*regular : 1)*e_s[h,b,e,s]
    //    (M=D, N=E, K=S; B accessed transposed: sb_k=1, sb_n=S)
    for (int b = 0; b < B; ++b)
        for (int h = 0; h < H; ++h)
            launch_gemm(dim3(E / 64, D / 128), stream,
                        etcv + ((size_t)(b * H + h)) * D * E,
                        v_ws + (size_t)b * HD * S + (size_t)h * D * S,
                        e_s + ((size_t)(h * B + b)) * E * S,
                        nullptr,
                        (h == 0) ? (mask + (size_t)b * S) : nullptr,
                        (h == 0) ? regular : nullptr,
                        D, E, S, /*sa_m*/S, /*sa_k*/1, /*sb_k*/1, /*sb_n*/S, /*ldc*/E);

    // 4) etc_k[b] = re_w @ etc_v[b] + re_b   (M=D, N=E, K=HD)
    for (int b = 0; b < B; ++b)
        launch_gemm(dim3(E / 64, D / 128), stream,
                    etck + (size_t)b * D * E, re_w, etcv + (size_t)b * H * D * E,
                    re_b, nullptr, nullptr,
                    D, E, HD, HD, 1, E, 1, E);

    // 5) pre-softmax scores[b,h][e,s] = sum_d (etc_k[b][d,e]*mix[h,d]) * q[b][d,s]
    //    (M=E, N=S, K=D; A accessed transposed: sa_m=1, sa_k=E)
    for (int b = 0; b < B; ++b)
        for (int h = 0; h < H; ++h)
            launch_gemm(dim3(S / 64, E / 128), stream,
                        sc + ((size_t)(b * H + h)) * E * S,
                        etck + (size_t)b * D * E, q_ws + (size_t)b * D * S,
                        nullptr, mix + (size_t)h * D, nullptr,
                        E, S, D, /*sa_m*/1, /*sa_k*/E, /*sb_k*/S, /*sb_n*/1, /*ldc*/S);

    // 6) softmax over E with temperature mask[b,s]*regular/sqrt_p (in place)
    softmax_etc<<<dim3(S / 8, B * H), dim3(256), 0, stream>>>(sc, mask, regular, sqrt_p);

    // 7) out1[b][h*D+d, s] = sum_e etc_v[b,h][d,e] * atten[b,h][e,s]
    for (int b = 0; b < B; ++b)
        for (int h = 0; h < H; ++h)
            launch_gemm(dim3(S / 64, D / 128), stream,
                        out1 + (size_t)b * HD * S + (size_t)h * D * S,
                        etcv + ((size_t)(b * H + h)) * D * E,
                        sc + ((size_t)(b * H + h)) * E * S,
                        nullptr, nullptr, nullptr,
                        D, S, E, E, 1, S, 1, S);

    // 8) out1 += avgpool3_same(multi_q)
    {
        long total = (long)B * H * D * S;
        pool_add<<<dim3((unsigned)((total + 255) / 256)), dim3(256), 0, stream>>>(
            out1, q_ws, mix, mask, regular);
    }

    // 9) final[b] = sq_w @ out1[b] + sq_b   (M=D, N=S, K=HD)
    for (int b = 0; b < B; ++b)
        launch_gemm(dim3(S / 64, D / 128), stream,
                    outF + (size_t)b * D * S, sq_w, out1 + (size_t)b * HD * S,
                    sq_b, nullptr, nullptr,
                    D, S, HD, HD, 1, S, 1, S);
}
